// ARONA_22067541967605
// MI455X (gfx1250) — matrix operand order repaired and checked by reading, not compiled, not run on hardware
//
#include <hip/hip_runtime.h>
#include <cstddef>
#include <cstdint>

// ---------------------------------------------------------------------------
// CDNA5 (gfx1250) GPT-2 forward pass.  wave32, WMMA bf16 16x16x32, f32 accum,
// double-buffered async global->LDS DMA staging (ASYNCcnt), flash attention.
// ---------------------------------------------------------------------------

typedef __attribute__((ext_vector_type(16))) __bf16 v16bf;
typedef __attribute__((ext_vector_type(8)))  float  v8f;
typedef __attribute__((ext_vector_type(4)))  float  f32x4;
typedef __attribute__((ext_vector_type(4)))  unsigned int u32x4;

union BfFrag { v16bf v; u32x4 u[2]; };

// --- bf16 fragment loader from bf16 LDS (used for the P restripe) ----------
static __device__ inline v16bf load_a_frag_bf16(const __bf16* row, int hi) {
  BfFrag f;
  f.u[0] = *(const u32x4*)(row + hi * 8);
  f.u[1] = *(const u32x4*)(row + 16 + hi * 8);
  return f.v;
}

// --- f32 -> bf16 fragment builders -----------------------------------------
static __device__ inline v16bf cvt16(f32x4 a, f32x4 b, f32x4 c, f32x4 d) {
  v16bf t;
  #pragma unroll
  for (int i = 0; i < 4; ++i) {
    t[i]      = (__bf16)a[i];
    t[4 + i]  = (__bf16)b[i];
    t[8 + i]  = (__bf16)c[i];
    t[12 + i] = (__bf16)d[i];
  }
  return t;
}
// A-fragment (16x32): lane (l&15)=row, k = hi*8 + j (+8 for j>=8):
// two contiguous 8-float runs at hi*8 and 16+hi*8.
static __device__ inline v16bf load_a_frag_f32(const float* row, int hi) {
  const f32x4* p0 = (const f32x4*)(row + hi * 8);
  const f32x4* p1 = (const f32x4*)(row + 16 + hi * 8);
  return cvt16(p0[0], p0[1], p1[0], p1[1]);
}
static __device__ inline v16bf load_a_frag_f32_scaled(const float* row, int hi,
                                                      float s) {
  const f32x4* p0 = (const f32x4*)(row + hi * 8);
  const f32x4* p1 = (const f32x4*)(row + 16 + hi * 8);
  return cvt16(p0[0] * s, p0[1] * s, p1[0] * s, p1[1] * s);
}
// B-fragment (32x16): lane (l&15)=col, k = hi*8 + j (+8 for j>=8):
// two contiguous 8-float runs at hi*8 and 16+hi*8.  Works from LDS [n][k] rows or global K-major rows.
static __device__ inline v16bf load_b_frag_f32(const float* col, int hi) {
  const f32x4* p0 = (const f32x4*)(col + hi * 8);
  const f32x4* p1 = (const f32x4*)(col + 16 + hi * 8);
  return cvt16(p0[0], p0[1], p1[0], p1[1]);
}

static __device__ inline v8f wmma_bf16(v16bf a, v16bf b, v8f c) {
  return __builtin_amdgcn_wmma_f32_16x16x32_bf16(false, a, false, b,
                                                 (short)0, c, false, false);
}
static __device__ inline v8f zero8() {
  v8f z = {0.f, 0.f, 0.f, 0.f, 0.f, 0.f, 0.f, 0.f};
  return z;
}

// async DMA: 16 bytes global -> LDS, tracked with ASYNCcnt
static __device__ inline void async_b128(uint32_t lds_byte_off,
                                         const float* gaddr) {
  asm volatile("global_load_async_to_lds_b128 %0, %1, off"
               :: "v"(lds_byte_off), "v"(gaddr) : "memory");
}
static __device__ inline void wait_async0() {
  asm volatile("s_wait_asynccnt 0x0" ::: "memory");
}

// ---------------------------------------------------------------------------
// Generic GEMM:  C[M,N] = op(A[M,K] @ B + bias) (+ resid)
// flags bit0: B stored [N,K]  (lm-head / tied embedding)  -- else [K,N]
// flags bit1: exact GELU epilogue
// Block: 256 thr (8 waves), tile 64(M) x 128(N), K-step 32, double-buffered:
// DMA of tile k+1 overlaps WMMA of tile k.
// ---------------------------------------------------------------------------
__global__ __launch_bounds__(256) void gemm_wmma(
    const float* __restrict__ A, const float* __restrict__ B,
    const float* __restrict__ bias, const float* __restrict__ resid,
    float* __restrict__ C, int M, int N, int K, int flags) {
  const bool b_nk    = flags & 1;
  const bool do_gelu = flags & 2;

  __shared__ __attribute__((aligned(16))) float sA[2][64 * 32];
  __shared__ __attribute__((aligned(16))) float sB[2][128 * 32];

  const int tid  = threadIdx.x;
  const int wave = tid >> 5;
  const int lane = tid & 31;
  const int l15  = lane & 15;
  const int hi   = lane >> 4;

  const int m0 = blockIdx.y * 64;
  const int n0 = blockIdx.x * 128;
  const int wr = (wave & 1) * 32;   // wave row offset within tile
  const int wc = (wave >> 1) * 32;  // wave col offset within tile

  // A staging: 8 consecutive floats per thread (row ar, cols ac..ac+7)
  const int ar = tid >> 2;
  const int ac = (tid & 3) * 8;
  const uint32_t ldsA[2] = {
      (uint32_t)(uintptr_t)&sA[0][0] + (uint32_t)tid * 32u,
      (uint32_t)(uintptr_t)&sA[1][0] + (uint32_t)tid * 32u};
  // B staging (b_nk): 16 consecutive floats per thread (row bn of [N,K])
  const int bn  = tid >> 1;
  const int bk0 = (tid & 1) * 16;
  const uint32_t ldsB[2] = {
      (uint32_t)(uintptr_t)&sB[0][0] + (uint32_t)tid * 64u,
      (uint32_t)(uintptr_t)&sB[1][0] + (uint32_t)tid * 64u};
  const bool bn_ok = (n0 + bn) < N;
  // transpose-staging coordinates for the [K,N] case
  const int tk = tid >> 3;
  const int tn = (tid & 7) * 16;

  // per-thread advancing global pointers (k-step = +32 along K)
  const float* gA = A + (size_t)(m0 + ar) * K + ac;
  const float* gBnk = B + (size_t)(n0 + bn) * K + bk0;           // b_nk
  const float* gBkn = B + (size_t)tk * N + n0 + tn;              // [K,N]

  // lm-head tail tiles: zero-fill LDS rows whose global row is OOB (once)
  if (b_nk && !bn_ok) {
    #pragma unroll
    for (int j = 0; j < 16; ++j) {
      sB[0][bn * 32 + bk0 + j] = 0.f;
      sB[1][bn * 32 + bk0 + j] = 0.f;
    }
  }

  v8f c00 = zero8(), c01 = zero8(), c10 = zero8(), c11 = zero8();

  const int ksteps = K >> 5;

  // stage a tile for k-step `ks` into buffer `bf`
  auto stage = [&](int ks, int bf) {
    const float* a = gA + ks * 32;
    async_b128(ldsA[bf], a);
    async_b128(ldsA[bf] + 16u, a + 4);
    if (b_nk) {
      if (bn_ok) {
        const float* p = gBnk + ks * 32;
        async_b128(ldsB[bf],       p);
        async_b128(ldsB[bf] + 16u, p + 4);
        async_b128(ldsB[bf] + 32u, p + 8);
        async_b128(ldsB[bf] + 48u, p + 12);
      }
    } else {
      // transpose [K,N] -> LDS [n][k]: vector loads along n, scatter stores
      const float* p = gBkn + (size_t)(ks * 32) * N;
      f32x4 r0 = ((const f32x4*)p)[0];
      f32x4 r1 = ((const f32x4*)p)[1];
      f32x4 r2 = ((const f32x4*)p)[2];
      f32x4 r3 = ((const f32x4*)p)[3];
      float* sb = sB[bf];
      #pragma unroll
      for (int j = 0; j < 4; ++j) {
        sb[(tn + j)      * 32 + tk] = r0[j];
        sb[(tn + 4 + j)  * 32 + tk] = r1[j];
        sb[(tn + 8 + j)  * 32 + tk] = r2[j];
        sb[(tn + 12 + j) * 32 + tk] = r3[j];
      }
    }
  };

  stage(0, 0);
  wait_async0();
  __syncthreads();

  int buf = 0;
  for (int ks = 0; ks < ksteps; ++ks) {
    if (ks + 1 < ksteps) stage(ks + 1, buf ^ 1);  // DMA next tile (overlap)

    const float* sa = sA[buf];
    const float* sb = sB[buf];
    v16bf a0 = load_a_frag_f32(&sa[(wr + l15) * 32], hi);
    v16bf a1 = load_a_frag_f32(&sa[(wr + 16 + l15) * 32], hi);
    v16bf b0 = load_b_frag_f32(&sb[(wc + l15) * 32], hi);
    v16bf b1 = load_b_frag_f32(&sb[(wc + 16 + l15) * 32], hi);
    c00 = wmma_bf16(a0, b0, c00);
    c01 = wmma_bf16(a0, b1, c01);
    c10 = wmma_bf16(a1, b0, c10);
    c11 = wmma_bf16(a1, b1, c11);

    wait_async0();     // next tile's DMA complete (this wave)
    __syncthreads();   // all waves done reading current tile
    buf ^= 1;
  }

  v8f acc[4] = {c00, c01, c10, c11};
  #pragma unroll
  for (int fr = 0; fr < 2; ++fr) {
    #pragma unroll
    for (int fc = 0; fc < 2; ++fc) {
      v8f a = acc[fr * 2 + fc];
      int col = n0 + wc + fc * 16 + l15;
      if (col < N) {
        float bvv = bias ? bias[col] : 0.f;
        #pragma unroll
        for (int r = 0; r < 8; ++r) {
          int row = m0 + wr + fr * 16 + hi * 8 + r;
          float vv = a[r] + bvv;
          if (do_gelu) vv = 0.5f * vv * (1.0f + erff(vv * 0.70710678118f));
          if (resid) vv += resid[(size_t)row * N + col];
          C[(size_t)row * N + col] = vv;
        }
      }
    }
  }
}

// ---------------------------------------------------------------------------
// Flash-style causal attention.  One wave per 16-query tile of one (b,h).
// Scores via WMMA (Q pre-scaled), online softmax with width-16 shfl_xor,
// P restriped via per-wave LDS (bf16), then P @ V via WMMA.
// Q, K, V, O layout: [B, S, H*64].
// ---------------------------------------------------------------------------
__global__ __launch_bounds__(256) void attn_wmma(
    const float* __restrict__ Q, const float* __restrict__ Kt,
    const float* __restrict__ Vt, const int* __restrict__ ids,
    float* __restrict__ O) {
  __shared__ __attribute__((aligned(16))) __bf16 sP[8][16 * 32];

  const int tid  = threadIdx.x;
  const int wave = tid >> 5;
  const int lane = tid & 31;
  const int l15  = lane & 15;
  const int hi   = lane >> 4;

  const int gid   = blockIdx.x * 8 + wave;
  const int qt    = gid & 63;          // S/16 = 64 query tiles
  const int h     = (gid >> 6) % 12;
  const int b     = gid / (64 * 12);
  const int qbase = qt * 16;

  const size_t bh = (size_t)b * 1024 * 768 + (size_t)h * 64;

  // Q A-fragments: k-runs are contiguous in d -> float4 loads, scale folded
  v16bf aq[2];
  #pragma unroll
  for (int kk = 0; kk < 2; ++kk) {
    const float* qr = Q + bh + (size_t)(qbase + l15) * 768 + kk * 32;
    aq[kk] = load_a_frag_f32_scaled(qr, hi, 0.125f);  // 1/sqrt(64)
  }

  v8f o0 = zero8(), o1 = zero8(), o2 = zero8(), o3 = zero8();
  float mrun[8], lrun[8];
  #pragma unroll
  for (int r = 0; r < 8; ++r) { mrun[r] = -1e30f; lrun[r] = 0.f; }

  const int nkb = (qbase + 16 + 31) >> 5;
  for (int jb = 0; jb < nkb; ++jb) {
    const int j0 = jb * 32;
    v8f s0 = zero8(), s1 = zero8();

    // scores: B-fragment k-run contiguous in d -> float4 loads from K rows
    #pragma unroll
    for (int kk = 0; kk < 2; ++kk) {
      #pragma unroll
      for (int ch = 0; ch < 2; ++ch) {
        const int n = j0 + ch * 16 + l15;  // key position
        const float* kp = Kt + bh + (size_t)n * 768 + kk * 32;
        v16bf bk = load_b_frag_f32(kp, hi);
        if (ch == 0) s0 = wmma_bf16(aq[kk], bk, s0);
        else         s1 = wmma_bf16(aq[kk], bk, s1);
      }
    }

    // causal + pad mask (C layout: row = r + hi*8, col = lane&15)
    const int c0 = j0 + l15, c1 = j0 + 16 + l15;
    const bool pad0 = (ids[b * 1024 + c0] != 0);
    const bool pad1 = (ids[b * 1024 + c1] != 0);
    #pragma unroll
    for (int r = 0; r < 8; ++r) {
      int qpos = qbase + r + hi * 8;
      if (!((c0 <= qpos) && pad0)) s0[r] = -1e30f;
      if (!((c1 <= qpos) && pad1)) s1[r] = -1e30f;
    }

    // online softmax (row reductions across 16 lanes of each half)
    float al[8];
    #pragma unroll
    for (int r = 0; r < 8; ++r) {
      float t = fmaxf(s0[r], s1[r]);
      #pragma unroll
      for (int off = 1; off < 16; off <<= 1)
        t = fmaxf(t, __shfl_xor(t, off, 16));
      float mn = fmaxf(mrun[r], t);
      al[r] = __expf(mrun[r] - mn);
      mrun[r] = mn;
    }
    #pragma unroll
    for (int r = 0; r < 8; ++r) {
      float p0 = __expf(s0[r] - mrun[r]);
      float p1 = __expf(s1[r] - mrun[r]);
      s0[r] = p0; s1[r] = p1;
      float u = p0 + p1;
      #pragma unroll
      for (int off = 1; off < 16; off <<= 1)
        u += __shfl_xor(u, off, 16);
      lrun[r] = lrun[r] * al[r] + u;
    }
    #pragma unroll
    for (int r = 0; r < 8; ++r) {
      o0[r] *= al[r]; o1[r] *= al[r]; o2[r] *= al[r]; o3[r] *= al[r];
    }

    // restripe P (C-layout) -> A-fragment via per-wave LDS scratch
    __bf16* pp = sP[wave];
    #pragma unroll
    for (int r = 0; r < 8; ++r) {
      pp[(r + hi * 8) * 32 + l15]      = (__bf16)s0[r];
      pp[(r + hi * 8) * 32 + 16 + l15] = (__bf16)s1[r];
    }
    asm volatile("s_wait_dscnt 0x0" ::: "memory");  // wave-local LDS RAW fence
    v16bf pf = load_a_frag_bf16(&pp[l15 * 32], hi);

    // O += P (16x32) @ V (32x64); V gather (stride 768 along k)
    #pragma unroll
    for (int nb = 0; nb < 4; ++nb) {
      v16bf bvf;
      const int n = nb * 16 + l15;  // output dim within head
      #pragma unroll
      for (int j = 0; j < 16; ++j) {
        int key = j0 + hi * 8 + j + (j & 8);
        bvf[j] = (__bf16)Vt[bh + (size_t)key * 768 + n];
      }
      if      (nb == 0) o0 = wmma_bf16(pf, bvf, o0);
      else if (nb == 1) o1 = wmma_bf16(pf, bvf, o1);
      else if (nb == 2) o2 = wmma_bf16(pf, bvf, o2);
      else              o3 = wmma_bf16(pf, bvf, o3);
    }
  }

  #pragma unroll
  for (int r = 0; r < 8; ++r) {
    float inv = lrun[r] > 0.f ? 1.0f / lrun[r] : 0.f;
    int q = qbase + r + hi * 8;
    size_t ob = bh + (size_t)q * 768;
    O[ob + l15]      = o0[r] * inv;
    O[ob + 16 + l15] = o1[r] * inv;
    O[ob + 32 + l15] = o2[r] * inv;
    O[ob + 48 + l15] = o3[r] * inv;
  }
}

// ---------------------------------------------------------------------------
// Elementwise / reduction kernels
// ---------------------------------------------------------------------------
__global__ void embed_k(const int* __restrict__ ids,
                        const float* __restrict__ emb,
                        const float* __restrict__ pos,
                        float* __restrict__ x) {
  const int t  = blockIdx.x;
  const int id = ids[t];
  const int s  = t & 1023;
  for (int c = threadIdx.x; c < 768; c += 256)
    x[(size_t)t * 768 + c] = emb[(size_t)id * 768 + c] + pos[s * 768 + c];
}

static __device__ inline float block_sum(float v, float* red, int tid) {
  red[tid] = v; __syncthreads();
  for (int off = 128; off > 0; off >>= 1) {
    if (tid < off) red[tid] += red[tid + off];
    __syncthreads();
  }
  float r = red[0]; __syncthreads();
  return r;
}

// Y = layernorm(X) * g + bsh   (rows of 768, block of 256)
__global__ __launch_bounds__(256) void layernorm_k(
    const float* __restrict__ X, const float* __restrict__ g,
    const float* __restrict__ bsh, float* __restrict__ Y) {
  __shared__ float red[256];
  const int row = blockIdx.x, tid = threadIdx.x;
  const float* xr = X + (size_t)row * 768;
  float v0 = xr[tid], v1 = xr[tid + 256], v2 = xr[tid + 512];
  float mean = block_sum(v0 + v1 + v2, red, tid) * (1.0f / 768.0f);
  float d0 = v0 - mean, d1 = v1 - mean, d2 = v2 - mean;
  float var = block_sum(d0 * d0 + d1 * d1 + d2 * d2, red, tid) * (1.0f / 768.0f);
  float rstd = rsqrtf(var + 1e-5f);
  float* yr = Y + (size_t)row * 768;
  yr[tid]       = d0 * rstd * g[tid]       + bsh[tid];
  yr[tid + 256] = d1 * rstd * g[tid + 256] + bsh[tid + 256];
  yr[tid + 512] = d2 * rstd * g[tid + 512] + bsh[tid + 512];
}

// X += layernorm(T) * g + bsh   (FFN trailing LN fused with residual add)
__global__ __launch_bounds__(256) void ln_residual_k(
    const float* __restrict__ T, const float* __restrict__ g,
    const float* __restrict__ bsh, float* __restrict__ X) {
  __shared__ float red[256];
  const int row = blockIdx.x, tid = threadIdx.x;
  const float* tr = T + (size_t)row * 768;
  float v0 = tr[tid], v1 = tr[tid + 256], v2 = tr[tid + 512];
  float mean = block_sum(v0 + v1 + v2, red, tid) * (1.0f / 768.0f);
  float d0 = v0 - mean, d1 = v1 - mean, d2 = v2 - mean;
  float var = block_sum(d0 * d0 + d1 * d1 + d2 * d2, red, tid) * (1.0f / 768.0f);
  float rstd = rsqrtf(var + 1e-5f);
  float* xr = X + (size_t)row * 768;
  xr[tid]       += d0 * rstd * g[tid]       + bsh[tid];
  xr[tid + 256] += d1 * rstd * g[tid + 256] + bsh[tid + 256];
  xr[tid + 512] += d2 * rstd * g[tid + 512] + bsh[tid + 512];
}

// per-row log-sum-exp + NLL over V=50257 logits
__global__ __launch_bounds__(256) void nll_k(
    const float* __restrict__ logits, const int* __restrict__ target,
    float* __restrict__ nll, float* __restrict__ msk) {
  __shared__ float red[256];
  const int row = blockIdx.x, tid = threadIdx.x;
  const float* lr = logits + (size_t)row * 50257;
  float mx = -1e30f;
  for (int i = tid; i < 50257; i += 256) mx = fmaxf(mx, lr[i]);
  red[tid] = mx; __syncthreads();
  for (int off = 128; off > 0; off >>= 1) {
    if (tid < off) red[tid] = fmaxf(red[tid], red[tid + off]);
    __syncthreads();
  }
  mx = red[0]; __syncthreads();
  float se = 0.f;
  for (int i = tid; i < 50257; i += 256) se += expf(lr[i] - mx);
  se = block_sum(se, red, tid);
  if (tid == 0) {
    float lse = mx + logf(se);
    int t = target[row];
    float valid = (t != 0) ? 1.f : 0.f;
    nll[row] = valid * (lse - lr[t]);
    msk[row] = valid;
  }
}

__global__ __launch_bounds__(256) void loss_k(
    const float* __restrict__ nll, const float* __restrict__ msk,
    float* __restrict__ out) {
  __shared__ float rn[256], rm[256];
  const int tid = threadIdx.x;
  float a = 0.f, c = 0.f;
  for (int i = tid; i < 4096; i += 256) { a += nll[i]; c += msk[i]; }
  rn[tid] = a; rm[tid] = c; __syncthreads();
  for (int off = 128; off > 0; off >>= 1) {
    if (tid < off) { rn[tid] += rn[tid + off]; rm[tid] += rm[tid + off]; }
    __syncthreads();
  }
  if (tid == 0) out[0] = rn[0] / fmaxf(rm[0], 1.0f);
}

// ---------------------------------------------------------------------------
// Orchestration
// ---------------------------------------------------------------------------
extern "C" void kernel_launch(void* const* d_in, const int* in_sizes, int n_in,
                              void* d_out, int out_size, void* d_ws,
                              size_t ws_size, hipStream_t stream) {
  (void)in_sizes; (void)n_in; (void)out_size; (void)ws_size;

  const int*   ids    = (const int*)  d_in[0];
  const int*   target = (const int*)  d_in[1];
  const float* emb    = (const float*)d_in[2];
  const float* pos    = (const float*)d_in[3];
  const float* ln1_s  = (const float*)d_in[4];
  const float* ln1_b  = (const float*)d_in[5];
  const float* wq     = (const float*)d_in[6];
  const float* bq     = (const float*)d_in[7];
  const float* wk     = (const float*)d_in[8];
  const float* bk     = (const float*)d_in[9];
  const float* wv     = (const float*)d_in[10];
  const float* bv     = (const float*)d_in[11];
  const float* wo     = (const float*)d_in[12];
  const float* bo     = (const float*)d_in[13];
  const float* ln2_s  = (const float*)d_in[14];
  const float* ln2_b  = (const float*)d_in[15];
  const float* wu     = (const float*)d_in[16];
  const float* bu     = (const float*)d_in[17];
  const float* wd     = (const float*)d_in[18];
  const float* bd     = (const float*)d_in[19];
  const float* fln_s  = (const float*)d_in[20];
  const float* fln_b  = (const float*)d_in[21];
  const float* lnf_s  = (const float*)d_in[22];
  const float* lnf_b  = (const float*)d_in[23];

  const int M = 4096, D = 768, DFF = 3072, V = 50257, L = 12;

  float* ws = (float*)d_ws;
  size_t off = 0;
  float* x    = ws + off; off += (size_t)M * D;
  float* hbf  = ws + off; off += (size_t)M * D;
  float* qb   = ws + off; off += (size_t)M * D;
  float* kb   = ws + off; off += (size_t)M * D;
  float* vb   = ws + off; off += (size_t)M * D;
  float* ob   = ws + off; off += (size_t)M * D;
  float* fb   = ws + off; off += (size_t)M * DFF;
  float* tb   = ws + off; off += (size_t)M * D;
  float* nllb = ws + off; off += M;
  float* mskb = ws + off; off += M;

  float* logits = (float*)d_out;

  const dim3 blk(256);
  const dim3 gD(D / 128, M / 64);      // 768-wide GEMMs
  const dim3 gF(DFF / 128, M / 64);    // 3072-wide GEMM
  const dim3 gV((V + 127) / 128, M / 64);

  embed_k<<<M, blk, 0, stream>>>(ids, emb, pos, x);

  for (int l = 0; l < L; ++l) {
    const size_t wDD = (size_t)l * D * D;
    layernorm_k<<<M, blk, 0, stream>>>(x, ln1_s + l * D, ln1_b + l * D, hbf);
    gemm_wmma<<<gD, blk, 0, stream>>>(hbf, wq + wDD, bq + l * D, nullptr, qb, M, D, D, 0);
    gemm_wmma<<<gD, blk, 0, stream>>>(hbf, wk + wDD, bk + l * D, nullptr, kb, M, D, D, 0);
    gemm_wmma<<<gD, blk, 0, stream>>>(hbf, wv + wDD, bv + l * D, nullptr, vb, M, D, D, 0);
    attn_wmma<<<384, blk, 0, stream>>>(qb, kb, vb, ids, ob);
    gemm_wmma<<<gD, blk, 0, stream>>>(ob, wo + wDD, bo + l * D, x, x, M, D, D, 0);
    layernorm_k<<<M, blk, 0, stream>>>(x, ln2_s + l * D, ln2_b + l * D, hbf);
    gemm_wmma<<<gF, blk, 0, stream>>>(hbf, wu + (size_t)l * D * DFF, bu + l * DFF,
                                      nullptr, fb, M, DFF, D, 2 /*gelu*/);
    gemm_wmma<<<gD, blk, 0, stream>>>(fb, wd + (size_t)l * DFF * D, bd + l * D,
                                      nullptr, tb, M, D, DFF, 0);
    ln_residual_k<<<M, blk, 0, stream>>>(tb, fln_s + l * D, fln_b + l * D, x);
  }

  layernorm_k<<<M, blk, 0, stream>>>(x, lnf_s, lnf_b, hbf);
  // tied lm_head: logits = h @ emb^T  (emb is [V, D] == [N, K] layout)
  gemm_wmma<<<gV, blk, 0, stream>>>(hbf, emb, nullptr, nullptr, logits, M, V, D, 1);

  nll_k<<<M, blk, 0, stream>>>(logits, target, nllb, mskb);
  loss_k<<<1, blk, 0, stream>>>(nllb, mskb, logits + (size_t)M * V);
}
